// SequencePro_8177617731965
// MI455X (gfx1250) — compile-verified, matrix operand order checked by reading, not run on hardware
//
#include <hip/hip_runtime.h>

typedef __attribute__((ext_vector_type(16))) __bf16 v16bf;
typedef __attribute__((ext_vector_type(8)))  __bf16 v8bf;
typedef __attribute__((ext_vector_type(4)))  __bf16 v4bf;
typedef __attribute__((ext_vector_type(8)))  float  v8f;
typedef __attribute__((ext_vector_type(4)))  float  v4f;

#define B_     64
#define T_     128
#define D_     512
#define N4_    2048
#define ALPHA_ 20.0f
#define XROW   520      // padded LDS row stride (bf16 elems): 520*2B = 1040B -> conflict-free frag loads
#define DSLICE 64       // d-columns per block

__device__ __forceinline__ float sigmoid_f(float x) {
    return 1.0f / (1.0f + __expf(-x));
}
__device__ __forceinline__ float tanh_f(float x) {
    float e = __expf(-2.0f * fabsf(x));
    float r = (1.0f - e) / (1.0f + e);
    return copysignf(r, x);
}

// Pass 1: bf16 WMMA GEMM for gates i,g,o (f is dead: c0 = 0), fused gate math,
// writes ALPHA*h to out. Grid: (D_/DSLICE, T_), block: 384 threads = 12 waves.
// Wave w -> gate (w>>2), n-tile (w&3); computes all 4 M-tiles so each weight
// element is read from HBM exactly once per block.
__global__ __launch_bounds__(384) void lstm_gates_wmma(
    const float* __restrict__ x, const float* __restrict__ w,
    float* __restrict__ hout)
{
    __shared__ __bf16 xlds[B_ * XROW];          // x[:,t,:] as bf16, padded
    __shared__ float  glds[3 * B_ * DSLICE];    // staged gate values i,g,o

    const int t   = blockIdx.y;
    const int d0  = blockIdx.x * DSLICE;
    const int tid = threadIdx.x;

    // Stage x[:, t, :] -> LDS (fp32 -> bf16), coalesced b128 reads
    for (int idx = tid; idx < B_ * (D_ / 4); idx += 384) {
        const int b = idx >> 7;       // 128 float4 chunks per row
        const int c = idx & 127;
        v4f v = *(const v4f*)(x + ((size_t)b * T_ + t) * D_ + c * 4);
        v4bf bv;
        #pragma unroll
        for (int i = 0; i < 4; ++i) bv[i] = (__bf16)v[i];
        *(v4bf*)(xlds + b * XROW + c * 4) = bv;
    }
    __syncthreads();

    const int wv    = tid >> 5;
    const int lane  = tid & 31;
    const int gate  = wv >> 2;                    // 0:i  1:g  2:o
    const int ntile = wv & 3;
    const int gbase = (gate == 0) ? 0 : ((gate == 1) ? 1024 : 1536);
    const int hl    = lane & 15;
    const int khalf = (lane < 16) ? 0 : 8;        // per-lane K sub-offset (A/B frag layout)

    // B-matrix source row: w_ih[t][n][:] is contiguous along K — ideal for frags
    const float* wrow = w + ((size_t)t * N4_ + gbase + d0 + ntile * 16 + hl) * D_;

    v8f acc[4] = {};

    for (int k0 = 0; k0 < D_; k0 += 32) {
        // B fragment: 16 fp32 weights -> v16bf (halves 0..7 = K+{0..7}, 8..15 = K+{16..23})
        v4f b0 = *(const v4f*)(wrow + k0 + khalf);
        v4f b1 = *(const v4f*)(wrow + k0 + khalf + 4);
        v4f b2 = *(const v4f*)(wrow + k0 + khalf + 16);
        v4f b3 = *(const v4f*)(wrow + k0 + khalf + 20);
        v16bf bf;
        #pragma unroll
        for (int i = 0; i < 4; ++i) {
            bf[i]      = (__bf16)b0[i];
            bf[4 + i]  = (__bf16)b1[i];
            bf[8 + i]  = (__bf16)b2[i];
            bf[12 + i] = (__bf16)b3[i];
        }
        if (k0 + 64 < D_) __builtin_prefetch(wrow + k0 + 64, 0, 0);

        #pragma unroll
        for (int mt = 0; mt < 4; ++mt) {
            const __bf16* ap = xlds + (mt * 16 + hl) * XROW + k0 + khalf;
            v8bf alo = *(const v8bf*)(ap);        // ds_load_b128
            v8bf ahi = *(const v8bf*)(ap + 16);   // ds_load_b128
            v16bf af = __builtin_shufflevector(alo, ahi,
                0, 1, 2, 3, 4, 5, 6, 7, 8, 9, 10, 11, 12, 13, 14, 15);
            acc[mt] = __builtin_amdgcn_wmma_f32_16x16x32_bf16(
                false, af, false, bf, (short)0, acc[mt], false, false);
        }
    }

    // Spill gate tiles to LDS (C layout: VGPR r -> row r / r+8, lane&15 -> col)
    const int colbase = ntile * 16 + hl;
    const int rshift  = (lane < 16) ? 0 : 8;
    #pragma unroll
    for (int mt = 0; mt < 4; ++mt) {
        #pragma unroll
        for (int r = 0; r < 8; ++r) {
            glds[gate * (B_ * DSLICE) + (mt * 16 + r + rshift) * DSLICE + colbase]
                = acc[mt][r];
        }
    }
    __syncthreads();

    // Fused gate math: h = sigmoid(o)*tanh(sigmoid(i)*tanh(g)); store ALPHA*h
    for (int idx = tid; idx < B_ * DSLICE; idx += 384) {
        const int b  = idx >> 6;
        const int dd = idx & 63;
        float gi = glds[idx];
        float gg = glds[B_ * DSLICE + idx];
        float go = glds[2 * B_ * DSLICE + idx];
        float c  = sigmoid_f(gi) * tanh_f(gg);
        float h  = sigmoid_f(go) * tanh_f(c);
        hout[((size_t)b * T_ + t) * D_ + d0 + dd] = ALPHA_ * h;
    }
}

// Pass 2: in-place row softmax over D=512. One wave per (b,t) row, 16 vals/lane.
__global__ __launch_bounds__(256) void softmax_rows(float* __restrict__ buf)
{
    const int row  = blockIdx.x * 8 + (threadIdx.x >> 5);   // b*T + t
    const int lane = threadIdx.x & 31;
    float* p = buf + (size_t)row * D_ + lane * 16;

    v4f v[4];
    #pragma unroll
    for (int i = 0; i < 4; ++i) v[i] = *(const v4f*)(p + i * 4);

    float m = -3.4e38f;
    #pragma unroll
    for (int i = 0; i < 4; ++i)
        #pragma unroll
        for (int j = 0; j < 4; ++j) m = fmaxf(m, v[i][j]);
    #pragma unroll
    for (int off = 16; off > 0; off >>= 1)
        m = fmaxf(m, __shfl_xor(m, off, 32));

    float s = 0.0f;
    #pragma unroll
    for (int i = 0; i < 4; ++i)
        #pragma unroll
        for (int j = 0; j < 4; ++j) { v[i][j] = __expf(v[i][j] - m); s += v[i][j]; }
    #pragma unroll
    for (int off = 16; off > 0; off >>= 1)
        s += __shfl_xor(s, off, 32);

    const float inv = 1.0f / s;
    #pragma unroll
    for (int i = 0; i < 4; ++i) {
        v4f o = v[i] * inv;
        *(v4f*)(p + i * 4) = o;
    }
}

extern "C" void kernel_launch(void* const* d_in, const int* in_sizes, int n_in,
                              void* d_out, int out_size, void* d_ws, size_t ws_size,
                              hipStream_t stream) {
    (void)in_sizes; (void)n_in; (void)d_ws; (void)ws_size; (void)out_size;
    const float* x = (const float*)d_in[0];   // [64,128,512] fp32
    const float* w = (const float*)d_in[1];   // [128,2048,512] fp32
    float* out = (float*)d_out;               // [64,128,512] fp32

    dim3 grid1(D_ / DSLICE, T_);              // 8 x 128 = 1024 blocks
    lstm_gates_wmma<<<grid1, 384, 0, stream>>>(x, w, out);
    softmax_rows<<<(B_ * T_) / 8, 256, 0, stream>>>(out);
}
